// StationaryFactorizedNoise_541165879638
// MI455X (gfx1250) — matrix operand order repaired and checked by reading, not compiled, not run on hardware
//
#include <hip/hip_runtime.h>
#include <hip/hip_bf16.h>

// ---------------------------------------------------------------------------
// StationaryFactorizedNoise for MI455X (gfx1250, wave32, WMMA f16 16x16x32)
//
// Stage 1: per-block circular conv as GEMM  yb(768x512) = x(768x416) * H(416x512)
//          H chunks staged in LDS via global_load_async_to_lds_b128 (ASYNCcnt
//          double buffering); overlap-add via global_atomic_add_f32 into
//          Y[768][65536] (f32, ws)
// Stage 2: dewhiten as GEMM (out^T tiles)   D(16d x 16t) = A2(16d x Kc) * Y(Kc x 16t)
//          Y chunks staged in LDS the same way
// ---------------------------------------------------------------------------

typedef _Float16 v16h __attribute__((ext_vector_type(16)));
typedef _Float16 v8h  __attribute__((ext_vector_type(8)));
typedef float    v8f  __attribute__((ext_vector_type(8)));
typedef float    v4f  __attribute__((ext_vector_type(4)));
typedef int      v4i  __attribute__((ext_vector_type(4)));

#define NROWS   768      // SIZE*C
#define TPAD    65656    // T_SIM + T_KERNEL - 1
#define HOP     392      // 512 - 121 + 1
#define NB      168      // ceil(TPAD/HOP)
#define BLK     512
#define TOUT    65536
#define CCH     384
#define S2M1    120      // s2 - 1
#define K1PAD   416      // 392 padded to 13*32

// ws layout (bytes)
#define OFF_H   0                        // 512 f32
#define OFF_B1  4096                     // 416*512 f16  (~426 KB)
#define OFF_A2  458752                   // 384*384 f16  (~295 KB)
#define OFF_Y   (1u<<20)                 // 768*65536 f32 (~201 MB)

#define AS1 __attribute__((address_space(1)))
#define AS3 __attribute__((address_space(3)))

#if __has_builtin(__builtin_amdgcn_global_load_async_to_lds_b128) && \
    __has_builtin(__builtin_amdgcn_s_wait_asynccnt)
#define USE_ASYNC 1
// prototype (from clang diagnostic): (as1 int4* gsrc, as3 int4* lds_dst, imm off, imm cpol)
#define ASYNC_CP16(gp, lp)                                              \
  __builtin_amdgcn_global_load_async_to_lds_b128(                       \
      (AS1 v4i*)(gp), (AS3 v4i*)(lp), 0, 0)
#define ASYNC_WAIT(n) __builtin_amdgcn_s_wait_asynccnt(n)
#else
#define USE_ASYNC 0
#define ASYNC_CP16(gp, lp) (*(v4f*)(lp) = *(const v4f*)(gp))
#define ASYNC_WAIT(n) ((void)0)
#endif

__device__ inline v16h mk16(v8h lo, v8h hi) {
  v16h r;
#pragma unroll
  for (int i = 0; i < 8; ++i) { r[i] = lo[i]; r[i + 8] = hi[i]; }
  return r;
}

__device__ inline v8h ld8_f32_h(const float* p, bool ok) {
  v8h r;
  if (ok) {
    v4f a = *(const v4f*)p;
    v4f b = *(const v4f*)(p + 4);
#pragma unroll
    for (int i = 0; i < 4; ++i) { r[i] = (_Float16)a[i]; r[i + 4] = (_Float16)b[i]; }
  } else {
#pragma unroll
    for (int i = 0; i < 8; ++i) r[i] = (_Float16)0.f;
  }
  return r;
}

// --- setup kernels ---------------------------------------------------------

// h = irfft(kernel_fft, 512, norm='backward') ; kernel_fft is real
__global__ void build_h(const float* __restrict__ kf, float* __restrict__ h) {
  int j = threadIdx.x;  // 512 threads
  float s = kf[0] + kf[256] * ((j & 1) ? -1.f : 1.f);
  for (int k = 1; k < 256; ++k) {
    int ph = (j * k) & 511;
    s += 2.f * kf[k] * cosf(6.283185307179586f * (float)ph * (1.f / 512.f));
  }
  h[j] = s * (1.f / 512.f);
}

// H[u][t] = (u<392) ? h[(t-u) mod 512] : 0   -- 416x512 f16 row-major
__global__ void build_b1(const float* __restrict__ h, _Float16* __restrict__ B1) {
  int idx = blockIdx.x * 256 + threadIdx.x;
  if (idx >= K1PAD * BLK) return;
  int u = idx >> 9, t = idx & 511;
  float v = (u < HOP) ? h[(t - u + 512) & 511] : 0.f;
  B1[idx] = (_Float16)v;
}

// A2[d][c] = std[c] * vt[c][d]   -- 384x384 f16 row-major (c contiguous)
__global__ void build_a2(const float* __restrict__ stdv, const float* __restrict__ vt,
                         _Float16* __restrict__ A2) {
  int idx = blockIdx.x * 256 + threadIdx.x;
  if (idx >= CCH * CCH) return;
  int d = idx / CCH, c = idx - d * CCH;
  A2[idx] = (_Float16)(stdv[c] * vt[c * CCH + d]);
}

__global__ void zero_y(v4f* __restrict__ Y) {
  v4f z; z[0] = 0.f; z[1] = 0.f; z[2] = 0.f; z[3] = 0.f;
  Y[(size_t)blockIdx.x * 256 + threadIdx.x] = z;
}

// --- stage 1: temporal filter GEMM + overlap-add ---------------------------
// grid: (NB=168, 24) ; block 256 = 8 waves
// wave -> mtile = blockIdx.y*2 + (wave>>2) (16 rows), N-quarter q = wave&3
#define B1W 520  // 512 + 8 halfs pad -> 1040 B row stride (bank-conflict free)
__global__ __launch_bounds__(256) void gemm1(const float* __restrict__ noise,
                                             const _Float16* __restrict__ B1,
                                             float* __restrict__ Y) {
  __shared__ _Float16 sB[2][32 * B1W];   // 2 x 33,280 B double buffer

  const int tid  = threadIdx.x;
  const int lane = tid & 31;
  const int wave = tid >> 5;
  const int b    = blockIdx.x;
  const int mbase = (blockIdx.y * 2 + (wave >> 2)) * 16;
  const int nq    = (wave & 3) * 8;             // first of 8 N-tiles
  const int klo   = (lane >> 4) * 8;            // A octet select per lane-half
  const float* rowp = noise + (size_t)(mbase + (lane & 15)) * TPAD;
  const int cbase = b * HOP;

  const int cr = tid >> 3, cs = tid & 7;        // cooperative-copy row / seg

  v8f acc[8];
#pragma unroll
  for (int j = 0; j < 8; ++j)
#pragma unroll
    for (int r = 0; r < 8; ++r) acc[j][r] = 0.f;

  // prefetch K-chunk 0 (32 rows x 512 halfs; 8 x b128 per thread)
#pragma unroll
  for (int i = 0; i < 8; ++i) {
    int seg = cs + i * 8;
    ASYNC_CP16(B1 + (size_t)cr * BLK + seg * 8, &sB[0][cr * B1W + seg * 8]);
  }

#pragma unroll 1
  for (int ks = 0; ks < 13; ++ks) {
    const int kb = ks * 32;
    if (ks + 1 < 13) {            // prefetch next chunk into other buffer
      const int kbn = kb + 32, bn = (ks + 1) & 1;
#pragma unroll
      for (int i = 0; i < 8; ++i) {
        int seg = cs + i * 8;
        ASYNC_CP16(B1 + (size_t)(kbn + cr) * BLK + seg * 8,
                   &sB[bn][cr * B1W + seg * 8]);
      }
      ASYNC_WAIT(8);              // in-order: chunk ks landed, ks+1 may fly
    } else {
      ASYNC_WAIT(0);
    }
    __syncthreads();

    // A fragment: global f32 -> f16 with pad/edge guards
    const int u0 = kb + klo, u1 = kb + klo + 16;
    v8h alo = ld8_f32_h(rowp + cbase + u0, (u0 < HOP) && (cbase + u0 < TPAD));
    v8h ahi = ld8_f32_h(rowp + cbase + u1, (u1 < HOP) && (cbase + u1 < TPAD));
    v16h a = mk16(alo, ahi);

    // phase 1: batch all 128 strided 16-bit LDS reads (one wait), phase 2: 8 WMMAs back-to-back
    const _Float16* bcol = &sB[ks & 1][lane & 15];   // B: lane = N col
    v8h blo[8], bhi[8];
#pragma unroll
    for (int j = 0; j < 8; ++j) {
#pragma unroll
      for (int s = 0; s < 8; ++s) {
        blo[j][s] = bcol[(klo + s) * B1W + (nq + j) * 16];
        bhi[j][s] = bcol[(klo + 16 + s) * B1W + (nq + j) * 16];
      }
    }
#pragma unroll
    for (int j = 0; j < 8; ++j) {
      v16h bf = mk16(blo[j], bhi[j]);
      acc[j] = __builtin_amdgcn_wmma_f32_16x16x32_f16(false, a, false, bf,
                                                      (short)0, acc[j], false, false);
    }
    __syncthreads();              // all reads done before buffer reuse
  }

  // overlap-add epilogue: Y[row][m-120] += yb[t], guarded to slice [120,65656)
#pragma unroll
  for (int j = 0; j < 8; ++j) {
    const int t = (nq + j) * 16 + (lane & 15);
    const int m = b * HOP + t;
    if (m >= S2M1 && m < TPAD) {
      float* yp = Y + (size_t)(mbase + (lane >> 4) * 8) * TOUT + (m - S2M1);
#pragma unroll
      for (int r = 0; r < 8; ++r) atomicAdd(yp + (size_t)r * TOUT, acc[j][r]);
    }
  }
}

// --- stage 2: dewhitening GEMM (out^T tiles -> coalesced d stores) ---------
// grid: 2048 t-chunks of 64 ; block 256 = 8 waves ; wave -> 3 d-tiles (24 total)
#define YW 68   // 64 + 4 f32 pad -> 272 B row stride (bank-conflict free)
__global__ __launch_bounds__(256) void gemm2(const float* __restrict__ Y,
                                             const _Float16* __restrict__ A2,
                                             float* __restrict__ out) {
  __shared__ float sY[2][32 * YW];       // 2 x 8,704 B double buffer

  const int tid  = threadIdx.x;
  const int lane = tid & 31;
  const int wave = tid >> 5;
  const size_t ntb = (size_t)blockIdx.x * 64;   // global n*65536 + t base
  const int n  = (int)(ntb >> 16);
  const int tl = (int)(ntb & 65535);            // 65536 % 64 == 0
  const int d0 = wave * 48;                     // 3 M-tiles of 16
  const int klo = (lane >> 4) * 8;

  const int cr = tid >> 3, cs = tid & 7;        // copy row / seg (2 x b128 each)
  const float* ybase = Y + (size_t)n * CCH * TOUT + tl;

  v8f acc[3][4];
#pragma unroll
  for (int i = 0; i < 3; ++i)
#pragma unroll
    for (int j = 0; j < 4; ++j)
#pragma unroll
      for (int r = 0; r < 8; ++r) acc[i][j][r] = 0.f;

  // prefetch K-chunk 0 (32 rows x 64 f32)
#pragma unroll
  for (int i = 0; i < 2; ++i) {
    int seg = cs + i * 8;
    ASYNC_CP16(ybase + (size_t)cr * TOUT + seg * 4, &sY[0][cr * YW + seg * 4]);
  }

#pragma unroll 1
  for (int ks = 0; ks < 12; ++ks) {
    const int kb = ks * 32;
    if (ks + 1 < 12) {
      const int kbn = kb + 32, bn = (ks + 1) & 1;
#pragma unroll
      for (int i = 0; i < 2; ++i) {
        int seg = cs + i * 8;
        ASYNC_CP16(ybase + (size_t)(kbn + cr) * TOUT + seg * 4,
                   &sY[bn][cr * YW + seg * 4]);
      }
      ASYNC_WAIT(2);
    } else {
      ASYNC_WAIT(0);
    }
    __syncthreads();

    // A fragments first (global, L2-hot) so they overlap the ds reads
    v16h af[3];
#pragma unroll
    for (int i = 0; i < 3; ++i) {
      const _Float16* ar = A2 + (size_t)(d0 + i * 16 + (lane & 15)) * CCH + kb;
      af[i] = mk16(*(const v8h*)(ar + klo), *(const v8h*)(ar + klo + 16));
    }

    // B fragments: batch all LDS reads, then convert, then WMMA chain
    const float* ycol = &sY[ks & 1][lane & 15];
    v4f yv[4][4];
#pragma unroll
    for (int j = 0; j < 4; ++j) {
      const float* p = ycol + j * 16;
#pragma unroll
      for (int q = 0; q < 4; ++q)
#pragma unroll
        for (int e = 0; e < 4; ++e) {
          const int s = q * 4 + e;
          const int k = (s < 8) ? klo + s : klo + 8 + s;
          yv[j][q][e] = p[k * YW];
        }
    }
    v16h bf[4];
#pragma unroll
    for (int j = 0; j < 4; ++j) {
      v16h t;
#pragma unroll
      for (int q = 0; q < 4; ++q)
#pragma unroll
        for (int e = 0; e < 4; ++e) t[q * 4 + e] = (_Float16)yv[j][q][e];
      bf[j] = t;
    }
#pragma unroll
    for (int i = 0; i < 3; ++i)
#pragma unroll
      for (int j = 0; j < 4; ++j)
        acc[i][j] = __builtin_amdgcn_wmma_f32_16x16x32_f16(false, af[i], false, bf[j],
                                                           (short)0, acc[i][j], false, false);
    __syncthreads();
  }

  // store: 8 consecutive d per lane -> two b128 stores
#pragma unroll
  for (int i = 0; i < 3; ++i)
#pragma unroll
    for (int j = 0; j < 4; ++j) {
      const int t = tl + j * 16 + (lane & 15);
      const int d = d0 + i * 16 + (lane >> 4) * 8;
      float* op = out + ((size_t)n * TOUT + t) * CCH + d;
      v4f lo, hi;
#pragma unroll
      for (int e = 0; e < 4; ++e) { lo[e] = acc[i][j][e]; hi[e] = acc[i][j][e + 4]; }
      *(v4f*)op = lo;
      *(v4f*)(op + 4) = hi;
    }
}

// ---------------------------------------------------------------------------
extern "C" void kernel_launch(void* const* d_in, const int* in_sizes, int n_in,
                              void* d_out, int out_size, void* d_ws, size_t ws_size,
                              hipStream_t stream) {
  const float* noise = (const float*)d_in[0];
  const float* stdv  = (const float*)d_in[1];
  const float* vt    = (const float*)d_in[2];
  const float* kf    = (const float*)d_in[3];

  char* ws = (char*)d_ws;
  float*    h  = (float*)(ws + OFF_H);
  _Float16* B1 = (_Float16*)(ws + OFF_B1);
  _Float16* A2 = (_Float16*)(ws + OFF_A2);
  float*    Y  = (float*)(ws + OFF_Y);

  build_h<<<1, 512, 0, stream>>>(kf, h);
  build_b1<<<(K1PAD * BLK + 255) / 256, 256, 0, stream>>>(h, B1);
  build_a2<<<(CCH * CCH + 255) / 256, 256, 0, stream>>>(stdv, vt, A2);
  zero_y<<<(NROWS * TOUT / 4) / 256, 256, 0, stream>>>((v4f*)Y);
  gemm1<<<dim3(NB, 24), 256, 0, stream>>>(noise, B1, Y);
  gemm2<<<2048, 256, 0, stream>>>(Y, A2, (float*)d_out);
}
